// MotionAutoencoder_50886772523418
// MI455X (gfx1250) — matrix operand order repaired and checked by reading, not compiled, not run on hardware
//
#include <hip/hip_runtime.h>
#include <hip/hip_bf16.h>

typedef __attribute__((ext_vector_type(16))) __bf16 v16bf;
typedef __attribute__((ext_vector_type(8)))  __bf16 bf16x8;
typedef __attribute__((ext_vector_type(8)))  float  v8f;

#define B_  256
#define T_  512
#define I_  156
#define H_  128
#define L_  32
#define IP_ 160   // I padded to multiple of 32
#define G4_ 512   // 4*H

// ---------------------------------------------------------------------------
// WMMA helpers (CDNA5 wave32, v_wmma_f32_16x16x32_bf16)
// ---------------------------------------------------------------------------
__device__ __forceinline__ v8f wmma_bf16(v16bf a, v16bf b, v8f c) {
  return __builtin_amdgcn_wmma_f32_16x16x32_bf16(false, a, false, b,
                                                 (short)0, c, false, false);
}

// A-matrix fragment (16x32 bf16, M = lane%16):
// lanes 0-15 : K = kbase+{0..7}, kbase+{16..23}
// lanes 16-31: K = kbase+{8..15}, kbase+{24..31}
__device__ __forceinline__ v16bf ldA(const __bf16* buf, int stride, int kbase) {
  const int lane = threadIdx.x & 31;
  const __bf16* p = buf + (lane & 15) * stride + kbase + ((lane >> 4) << 3);
  bf16x8 lo = *(const bf16x8*)p;
  bf16x8 hi = *(const bf16x8*)(p + 16);
  v16bf a;
#pragma unroll
  for (int i = 0; i < 8; ++i) { a[i] = lo[i]; a[i + 8] = hi[i]; }
  return a;
}

// B-matrix fragment (32x16 bf16, column N = n0 + lane%16):
// lanes 0-15 : K = kbase+{0..7}, kbase+{16..23}; lanes 16-31: K = kbase+{8..15}, kbase+{24..31}.
// Weights stored row-major [N][K] in LDS -> two contiguous 8-elem reads per lane (at k offsets 0 and 16).
__device__ __forceinline__ v16bf ldB(const __bf16* buf, int stride, int n0, int kbase) {
  const int lane = threadIdx.x & 31;
  const __bf16* p = buf + (n0 + (lane & 15)) * stride + kbase + ((lane >> 4) << 3);
  bf16x8 lo = *(const bf16x8*)p;
  bf16x8 hi = *(const bf16x8*)(p + 16);
  v16bf b;
#pragma unroll
  for (int i = 0; i < 8; ++i) { b[i] = lo[i]; b[i + 8] = hi[i]; }
  return b;
}

__device__ __forceinline__ float sigf(float x) { return 1.0f / (1.0f + __expf(-x)); }

// ---------------------------------------------------------------------------
// Encoder LSTM scan: 16 WGs x 16 batch rows, weight-stationary in LDS.
// ---------------------------------------------------------------------------
__global__ void __launch_bounds__(256, 1)
vae_enc_scan(const float* __restrict__ motion, const float* __restrict__ Wih,
             const float* __restrict__ Whh, const float* __restrict__ bias,
             float* __restrict__ hfin) {
  extern __shared__ char smem[];
  __bf16* sWih = (__bf16*)smem;            // [512][160] bf16
  __bf16* sWhh = sWih + G4_ * IP_;         // [512][128] bf16
  __bf16* sX   = sWhh + G4_ * H_;          // [16][160]  bf16
  __bf16* sH   = sX + 16 * IP_;            // [16][128]  bf16
  float*  sB   = (float*)(sH + 16 * H_);   // [512] f32

  const int tid  = threadIdx.x;
  const int wave = tid >> 5;
  const int lane = tid & 31;
  const int nrow = lane & 15;
  const int rhi  = (lane >> 4) << 3;
  const int b0   = blockIdx.x * 16;

  for (int e = tid; e < G4_ * IP_; e += 256) {
    int n = e / IP_, k = e - n * IP_;
    sWih[e] = (k < I_) ? (__bf16)Wih[n * I_ + k] : (__bf16)0.0f;
  }
  for (int e = tid; e < G4_ * H_; e += 256) sWhh[e] = (__bf16)Whh[e];
  for (int e = tid; e < G4_; e += 256)      sB[e]   = bias[e];
  for (int e = tid; e < 16 * H_; e += 256)  sH[e]   = (__bf16)0.0f;
  __syncthreads();

  v8f cst, hreg;
#pragma unroll
  for (int r = 0; r < 8; ++r) { cst[r] = 0.0f; hreg[r] = 0.0f; }

  for (int t = 0; t < T_; ++t) {
    // stage x_t (f32 -> bf16) into LDS
    for (int e = tid; e < 16 * IP_; e += 256) {
      int br = e / IP_, i = e - br * IP_;
      float v = (i < I_) ? motion[((size_t)(b0 + br) * T_ + t) * I_ + i] : 0.0f;
      sX[e] = (__bf16)v;
    }
    if (tid < 16 && t + 1 < T_)
      __builtin_prefetch(motion + ((size_t)(b0 + tid) * T_ + (t + 1)) * I_, 0, 0);
    __syncthreads();   // sX staged, sH holds h_{t-1}

    // gates[16,512] = x@Wih^T + h@Whh^T + b ; wave w owns N-tiles {w,w+8,w+16,w+24}
    v8f acc[4];
#pragma unroll
    for (int g = 0; g < 4; ++g) {
      const int n0 = (g * 8 + wave) * 16;
      const float bv = sB[n0 + nrow];
#pragma unroll
      for (int r = 0; r < 8; ++r) acc[g][r] = bv;
#pragma unroll
      for (int kt = 0; kt < IP_ / 32; ++kt)
        acc[g] = wmma_bf16(ldA(sX, IP_, kt * 32), ldB(sWih, IP_, n0, kt * 32), acc[g]);
#pragma unroll
      for (int kt = 0; kt < H_ / 32; ++kt)
        acc[g] = wmma_bf16(ldA(sH, H_, kt * 32), ldB(sWhh, H_, n0, kt * 32), acc[g]);
    }
    __syncthreads();   // all reads of sX/sH complete

    // LSTM cell, f32 in registers (PyTorch gate order i,f,g,o)
#pragma unroll
    for (int r = 0; r < 8; ++r) {
      float ig = sigf(acc[0][r]);
      float fg = sigf(acc[1][r]);
      float gg = tanhf(acc[2][r]);
      float og = sigf(acc[3][r]);
      float c  = fg * cst[r] + ig * gg;
      cst[r]   = c;
      hreg[r]  = og * tanhf(c);
    }
    const int col = wave * 16 + nrow;
#pragma unroll
    for (int r = 0; r < 8; ++r) sH[(r + rhi) * H_ + col] = (__bf16)hreg[r];
  }

  const int col = wave * 16 + nrow;
#pragma unroll
  for (int r = 0; r < 8; ++r)
    hfin[(size_t)(b0 + r + rhi) * H_ + col] = hreg[r];
}

// ---------------------------------------------------------------------------
// Latent head: mu, logvar, z, KL partial sums.
// ---------------------------------------------------------------------------
__global__ void vae_latent(const float* __restrict__ hfin, const float* __restrict__ eps,
                           const float* __restrict__ muW, const float* __restrict__ mub,
                           const float* __restrict__ vaW, const float* __restrict__ vab,
                           float* __restrict__ out_mu, float* __restrict__ out_lv,
                           float* __restrict__ z, float* __restrict__ kl_acc) {
  const int idx = blockIdx.x * blockDim.x + threadIdx.x;  // B*L = 8192
  const int b = idx >> 5, l = idx & 31;
  const float* h = hfin + (size_t)b * H_;
  float m = mub[l], v = vab[l];
#pragma unroll 4
  for (int k = 0; k < H_; ++k) {
    m += h[k] * muW[l * H_ + k];
    v += h[k] * vaW[l * H_ + k];
  }
  out_mu[idx] = m;
  out_lv[idx] = v;
  z[idx] = m + eps[idx] * __expf(0.5f * v);
  float kc = -0.5f * (1.0f + v - m * m - __expf(v));
#pragma unroll
  for (int off = 16; off > 0; off >>= 1) kc += __shfl_down(kc, off, 32);
  if ((threadIdx.x & 31) == 0) atomicAdd(kl_acc, kc);
}

// dh0 = z @ dec_fc_W^T + dec_fc_b
__global__ void vae_dh0(const float* __restrict__ z, const float* __restrict__ fcW,
                        const float* __restrict__ fcb, float* __restrict__ dh0) {
  const int idx = blockIdx.x * blockDim.x + threadIdx.x;  // B*H = 32768
  const int b = idx >> 7, hh = idx & 127;
  float s = fcb[hh];
#pragma unroll
  for (int l = 0; l < L_; ++l) s += z[(size_t)b * L_ + l] * fcW[hh * L_ + l];
  dh0[idx] = s;
}

// ---------------------------------------------------------------------------
// Decoder LSTM scan with fused output projection + rec-loss accumulation.
// gates = h @ (dec_Wih + dec_Whh)^T + b   (same h feeds both in the reference)
// ---------------------------------------------------------------------------
__global__ void __launch_bounds__(256, 1)
vae_dec_scan(const float* __restrict__ motion, const float* __restrict__ dh0,
             const float* __restrict__ dWih, const float* __restrict__ dWhh,
             const float* __restrict__ dbias, const float* __restrict__ outW,
             const float* __restrict__ outb, float* __restrict__ recon,
             float* __restrict__ rec_acc) {
  extern __shared__ char smem[];
  __bf16* sW  = (__bf16*)smem;           // [512][128] bf16 (Wih+Whh folded)
  __bf16* sO  = sW + G4_ * H_;           // [160][128] bf16 (out_W, padded)
  __bf16* sH  = sO + IP_ * H_;           // [16][128]  bf16
  float*  sBd = (float*)(sH + 16 * H_);  // [512] f32
  float*  sOb = sBd + G4_;               // [160] f32

  const int tid  = threadIdx.x;
  const int wave = tid >> 5;
  const int lane = tid & 31;
  const int nrow = lane & 15;
  const int rhi  = (lane >> 4) << 3;
  const int b0   = blockIdx.x * 16;

  for (int e = tid; e < G4_ * H_; e += 256) sW[e] = (__bf16)(dWih[e] + dWhh[e]);
  for (int e = tid; e < IP_ * H_; e += 256) {
    int o = e / H_, k = e - o * H_;
    sO[e] = (o < I_) ? (__bf16)outW[o * H_ + k] : (__bf16)0.0f;
  }
  for (int e = tid; e < G4_; e += 256) sBd[e] = dbias[e];
  for (int e = tid; e < IP_; e += 256) sOb[e] = (e < I_) ? outb[e] : 0.0f;
  for (int e = tid; e < 16 * H_; e += 256) sH[e] = (__bf16)dh0[(size_t)b0 * H_ + e];
  __syncthreads();

  v8f cst;
#pragma unroll
  for (int r = 0; r < 8; ++r) cst[r] = 0.0f;
  float recsum = 0.0f;

  for (int t = 0; t < T_; ++t) {
    // gates from h_{t-1} (resident in sH)
    v8f acc[4];
#pragma unroll
    for (int g = 0; g < 4; ++g) {
      const int n0 = (g * 8 + wave) * 16;
      const float bv = sBd[n0 + nrow];
#pragma unroll
      for (int r = 0; r < 8; ++r) acc[g][r] = bv;
#pragma unroll
      for (int kt = 0; kt < H_ / 32; ++kt)
        acc[g] = wmma_bf16(ldA(sH, H_, kt * 32), ldB(sW, H_, n0, kt * 32), acc[g]);
    }
    __syncthreads();   // reads of h_{t-1} done

    v8f hreg;
#pragma unroll
    for (int r = 0; r < 8; ++r) {
      float ig = sigf(acc[0][r]);
      float fg = sigf(acc[1][r]);
      float gg = tanhf(acc[2][r]);
      float og = sigf(acc[3][r]);
      float c  = fg * cst[r] + ig * gg;
      cst[r]   = c;
      hreg[r]  = og * tanhf(c);
    }
    const int col = wave * 16 + nrow;
#pragma unroll
    for (int r = 0; r < 8; ++r) sH[(r + rhi) * H_ + col] = (__bf16)hreg[r];
    __syncthreads();   // sH now holds h_t

    // recon_t = h_t @ out_W^T : 10 N-tiles, waves 0..7 take tile w; 0,1 also w+8
    const int ntiles = (wave < 2) ? 2 : 1;
    for (int s = 0; s < ntiles; ++s) {
      const int n0 = (wave + s * 8) * 16;
      v8f racc;
      const float obv = sOb[n0 + nrow];
#pragma unroll
      for (int r = 0; r < 8; ++r) racc[r] = obv;
#pragma unroll
      for (int kt = 0; kt < H_ / 32; ++kt)
        racc = wmma_bf16(ldA(sH, H_, kt * 32), ldB(sO, H_, n0, kt * 32), racc);

      const int o = n0 + nrow;
      if (o < I_) {
#pragma unroll
        for (int r = 0; r < 8; ++r) {
          const size_t idx = ((size_t)(b0 + r + rhi) * T_ + t) * I_ + o;
          const float rv = racc[r];
          recon[idx] = rv;
          const float d = rv - motion[idx];
          recsum += d * d;
        }
      }
    }
  }

#pragma unroll
  for (int off = 16; off > 0; off >>= 1) recsum += __shfl_down(recsum, off, 32);
  if (lane == 0) atomicAdd(rec_acc, recsum);
}

// ---------------------------------------------------------------------------
__global__ void vae_init_acc(float* wsf) {
  if (threadIdx.x < 2) wsf[threadIdx.x] = 0.0f;
}

__global__ void vae_finalize(const float* wsf, float* total) {
  *total = wsf[0] * (1.0f / ((float)B_ * (float)T_ * (float)I_)) + wsf[1];
}

// ---------------------------------------------------------------------------
extern "C" void kernel_launch(void* const* d_in, const int* in_sizes, int n_in,
                              void* d_out, int out_size, void* d_ws, size_t ws_size,
                              hipStream_t stream) {
  (void)in_sizes; (void)n_in; (void)out_size; (void)ws_size;
  const float* motion = (const float*)d_in[0];
  const float* eps    = (const float*)d_in[1];
  const float* eWih   = (const float*)d_in[2];
  const float* eWhh   = (const float*)d_in[3];
  const float* eb     = (const float*)d_in[4];
  const float* muW    = (const float*)d_in[5];
  const float* mub    = (const float*)d_in[6];
  const float* vaW    = (const float*)d_in[7];
  const float* vab    = (const float*)d_in[8];
  const float* fcW    = (const float*)d_in[9];
  const float* fcb    = (const float*)d_in[10];
  const float* dWih   = (const float*)d_in[11];
  const float* dWhh   = (const float*)d_in[12];
  const float* db     = (const float*)d_in[13];
  const float* oW     = (const float*)d_in[14];
  const float* ob     = (const float*)d_in[15];

  float* out      = (float*)d_out;
  float* recon    = out;                                   // [B,T,I]
  float* out_mu   = out + (size_t)B_ * T_ * I_;            // [B,L]
  float* out_lv   = out_mu + (size_t)B_ * L_;              // [B,L]
  float* out_tot  = out_lv + (size_t)B_ * L_;              // scalar

  float* wsf     = (float*)d_ws;
  float* rec_acc = wsf + 0;
  float* kl_acc  = wsf + 1;
  float* hfin    = wsf + 2;                                // [B,H]
  float* zbuf    = hfin + (size_t)B_ * H_;                 // [B,L]
  float* dh0     = zbuf + (size_t)B_ * L_;                 // [B,H]

  const size_t enc_lds = (size_t)(G4_ * IP_ + G4_ * H_ + 16 * IP_ + 16 * H_) * 2
                       + (size_t)G4_ * 4;                  // 306176 B (< 320 KB/WGP)
  const size_t dec_lds = (size_t)(G4_ * H_ + IP_ * H_ + 16 * H_) * 2
                       + (size_t)G4_ * 4 + (size_t)IP_ * 4; // 178816 B

  vae_init_acc<<<1, 32, 0, stream>>>(wsf);
  vae_enc_scan<<<16, 256, enc_lds, stream>>>(motion, eWih, eWhh, eb, hfin);
  vae_latent<<<(B_ * L_) / 256, 256, 0, stream>>>(hfin, eps, muW, mub, vaW, vab,
                                                  out_mu, out_lv, zbuf, kl_acc);
  vae_dh0<<<(B_ * H_) / 256, 256, 0, stream>>>(zbuf, fcW, fcb, dh0);
  vae_dec_scan<<<16, 256, dec_lds, stream>>>(motion, dh0, dWih, dWhh, db, oW, ob,
                                             recon, rec_acc);
  vae_finalize<<<1, 1, 0, stream>>>(wsf, out_tot);
}
